// LRU_18408229831103
// MI455X (gfx1250) — compile-verified, matrix operand order checked by reading, not run on hardware
//
#include <hip/hip_runtime.h>
#include <cstdint>
#include <cstddef>

// Problem sizes (match reference)
#define SEQ   4096
#define BATCH 32
#define DDIM  512
#define HDIM  512

typedef __attribute__((ext_vector_type(16))) __bf16 v16bf;
typedef __attribute__((ext_vector_type(8)))  float  v8f;

// A/B fragment for V_WMMA_F32_16X16X32_BF16: 16 bf16 = 8 VGPRs per lane.
union Frag {
  v16bf v;
  unsigned short u[16];
  unsigned int   d[8];
  uint4          q[2];
};

// float pair -> packed bf16x2.
#if __has_builtin(__builtin_amdgcn_cvt_pk_bf16_f32)
typedef __attribute__((ext_vector_type(2))) __bf16 v2bf;
__device__ __forceinline__ unsigned int pack_bf(float lo, float hi) {
  v2bf p = __builtin_amdgcn_cvt_pk_bf16_f32(lo, hi);
  return __builtin_bit_cast(unsigned int, p);
}
__device__ __forceinline__ unsigned short f2bf(float f) {
  return (unsigned short)(pack_bf(f, 0.0f) & 0xFFFFu);
}
#elif __has_builtin(__builtin_amdgcn_perm)
// Round-to-nearest (ties away) via +0x8000, then take the two high halves
// with a single v_perm_b32: bytes {uh.b3, uh.b2, ul.b3, ul.b2}.
__device__ __forceinline__ unsigned short f2bf(float f) {
  unsigned int u = __builtin_bit_cast(unsigned int, f);
  return (unsigned short)((u + 0x8000u) >> 16);
}
__device__ __forceinline__ unsigned int pack_bf(float lo, float hi) {
  unsigned int ul = __builtin_bit_cast(unsigned int, lo) + 0x8000u;
  unsigned int uh = __builtin_bit_cast(unsigned int, hi) + 0x8000u;
  return __builtin_amdgcn_perm(uh, ul, 0x07060302u);
}
#else
__device__ __forceinline__ unsigned short f2bf(float f) {
  unsigned int u = __builtin_bit_cast(unsigned int, f);
  return (unsigned short)((u + 0x8000u) >> 16);
}
__device__ __forceinline__ unsigned int pack_bf(float lo, float hi) {
  unsigned int ul = __builtin_bit_cast(unsigned int, lo) + 0x8000u;
  unsigned int uh = __builtin_bit_cast(unsigned int, hi) + 0x8000u;
  return (uh & 0xFFFF0000u) | (ul >> 16);
}
#endif

// Pack 8 consecutive floats (two float4) into one 16B bf16 chunk.
__device__ __forceinline__ uint4 pack_chunk(const float4 a, const float4 b) {
  uint4 r;
  r.x = pack_bf(a.x, a.y);
  r.y = pack_bf(a.z, a.w);
  r.z = pack_bf(b.x, b.y);
  r.w = pack_bf(b.z, b.w);
  return r;
}

// Build an A/B fragment from two 8-float global reads (row-major f32 source).
// ISA 7.12.2 16-bit layout: half 0 = K {0..7}+grp*8, half 1 = K {16..23}+grp*8.
__device__ __forceinline__ Frag load_frag_global(const float* __restrict__ row,
                                                 int kbase) {
  Frag f;
#pragma unroll
  for (int h2 = 0; h2 < 2; ++h2) {
    const int k = kbase + h2 * 16;
    const float4 f0 = *reinterpret_cast<const float4*>(&row[k]);
    const float4 f1 = *reinterpret_cast<const float4*>(&row[k + 4]);
    f.q[h2] = pack_chunk(f0, f1);
  }
  return f;
}

// ---------------------------------------------------------------------------
// Phase 1: xw[s*B+b][h] = sum_d x[s*B+b][d] * W_ih[h][d] + bias[h]
// M = 131072, N = 512, K = 512.  Block: 256 thr (8 waves), each wave one
// 16-row M tile; block covers 128 rows x 64 cols.  Memory-bound; bf16 WMMA.
// W slice lives in LDS as bf16 with XOR chunk swizzle -> conflict-free
// ds_load_b128 fragment reads.
// ---------------------------------------------------------------------------
__global__ __launch_bounds__(256) void rnn_proj_kernel(
    const float* __restrict__ x, const float* __restrict__ Wih,
    const float* __restrict__ bias, float* __restrict__ out) {
  // bf16 [n][k], 64*512*2 = 64 KB; 16B chunk c of row n stored at c ^ (n&15).
  __shared__ unsigned short Ws[64 * DDIM];

  const int tid  = threadIdx.x;
  const int lane = tid & 31;
  const int wave = tid >> 5;
  const int grp  = lane >> 4;
  const int lm   = lane & 15;
  const int mBase = blockIdx.x * 128 + wave * 16;
  const int nBase = blockIdx.y * 64;

  // Stage W_ih[nBase..nBase+63][:] -> LDS (f32 -> bf16, swizzled chunks).
  for (int c = tid; c < 64 * (DDIM / 8); c += 256) {
    const int n  = c >> 6;          // row within slice
    const int cl = c & 63;          // logical 8-elem chunk within row
    const float* src = &Wih[(size_t)(nBase + n) * DDIM + cl * 8];
    const float4 f0 = *reinterpret_cast<const float4*>(src);
    const float4 f1 = *reinterpret_cast<const float4*>(src + 4);
    *reinterpret_cast<uint4*>(&Ws[n * DDIM + ((cl ^ (n & 15)) << 3)]) =
        pack_chunk(f0, f1);
  }
  __syncthreads();

  v8f acc[4];
#pragma unroll
  for (int nt = 0; nt < 4; ++nt)
#pragma unroll
    for (int i = 0; i < 8; ++i) acc[nt][i] = 0.0f;

  const float* xrow = &x[(size_t)(mBase + lm) * DDIM];

#pragma unroll 4
  for (int kc = 0; kc < DDIM / 32; ++kc) {
    const Frag a = load_frag_global(xrow, kc * 32 + grp * 8);
#pragma unroll
    for (int nt = 0; nt < 4; ++nt) {
      const int n = nt * 16 + lm;
      Frag bb;
#pragma unroll
      for (int h2 = 0; h2 < 2; ++h2) {
        const int chunk = (kc * 32 + grp * 8 + h2 * 16) >> 3;
        bb.q[h2] = *reinterpret_cast<const uint4*>(
            &Ws[n * DDIM + ((chunk ^ (n & 15)) << 3)]);
      }
      acc[nt] = __builtin_amdgcn_wmma_f32_16x16x32_bf16(
          false, a.v, false, bb.v, (short)0, acc[nt], false, false);
    }
  }

  // Add bias, store f32 results (C/D layout: m = v + 8*grp, n = lm).
#pragma unroll
  for (int nt = 0; nt < 4; ++nt) {
    const int n  = nBase + nt * 16 + lm;
    const float bv = bias[n];
#pragma unroll
    for (int v = 0; v < 8; ++v) {
      const int m = mBase + v + grp * 8;
      out[(size_t)m * HDIM + n] = acc[nt][v] + bv;
    }
  }
}

// ---------------------------------------------------------------------------
// Phase 2: h_t = xw_t + h_{t-1} @ W_hh^T, in place over out[t] = xw_t.
// Batches are independent -> 2 sync-free blocks of 16 batches each.
// Block: 1024 thr = 32 waves; wave w owns N columns [16w, 16w+16) and keeps
// its W_hh B-fragments (16 x 8 VGPRs = 128 VGPRs) resident for all 4096
// steps.  h ping-pongs in LDS as bf16 (row pad 8 -> conflict-free b128
// fragment reads); one s_barrier per step.
// ---------------------------------------------------------------------------
#define HROW (HDIM + 8)
__global__ __launch_bounds__(1024) void rnn_scan_kernel(
    const float* __restrict__ Whh, float* __restrict__ out) {
  __shared__ unsigned short hbuf[2][16][HROW];  // ~33 KB

  const int tid  = threadIdx.x;
  const int lane = tid & 31;
  const int wave = tid >> 5;      // 0..31
  const int grp  = lane >> 4;
  const int lm   = lane & 15;
  const int g    = blockIdx.x;    // batch group: rows 16g .. 16g+15
  const int n0   = wave * 16;

  // Preload this wave's B operand: B[k][n] = W_hh[n][k], n in [n0, n0+16).
  Frag wf[16];
  {
    const float* wrow = &Whh[(size_t)(n0 + lm) * HDIM];
#pragma unroll
    for (int kc = 0; kc < 16; ++kc)
      wf[kc] = load_frag_global(wrow, kc * 32 + grp * 8);
  }

  // h_0 = 0
  for (int i = tid; i < 16 * HROW; i += 1024)
    (&hbuf[0][0][0])[i] = 0;
  __syncthreads();

  // Per-lane LDS base for A-fragment reads (row = lm) in each buffer.
  const unsigned short* aRow[2] = {&hbuf[0][lm][0], &hbuf[1][lm][0]};
  size_t obase = ((size_t)g * 16) * HDIM;

#pragma unroll 1
  for (int t = 0; t < SEQ; ++t) {
    const int cur = t & 1;
    const unsigned short* arow = aRow[cur];

    v8f acc;
#pragma unroll
    for (int i = 0; i < 8; ++i) acc[i] = 0.0f;

#pragma unroll
    for (int kc = 0; kc < 16; ++kc) {
      Frag a;  // A[m][k] = h_{t-1}[m][k], m = lm
#pragma unroll
      for (int h2 = 0; h2 < 2; ++h2) {
        a.q[h2] = *reinterpret_cast<const uint4*>(
            &arow[kc * 32 + grp * 8 + h2 * 16]);
      }
      acc = __builtin_amdgcn_wmma_f32_16x16x32_bf16(
          false, a.v, false, wf[kc].v, (short)0, acc, false, false);
    }

    // h_t = xw_t + acc; write back in place and stage bf16 copy for t+1.
    const int nxt = cur ^ 1;
#pragma unroll
    for (int v = 0; v < 8; ++v) {
      const int m = v + grp * 8;
      const int n = n0 + lm;
      const size_t o = obase + (size_t)m * HDIM + n;
      const float h = out[o] + acc[v];
      out[o] = h;
      hbuf[nxt][m][n] = f2bf(h);
    }
    obase += (size_t)BATCH * HDIM;
    __syncthreads();
  }
}

// ---------------------------------------------------------------------------
extern "C" void kernel_launch(void* const* d_in, const int* in_sizes, int n_in,
                              void* d_out, int out_size, void* d_ws,
                              size_t ws_size, hipStream_t stream) {
  (void)in_sizes; (void)n_in; (void)out_size; (void)d_ws; (void)ws_size;
  const float* x   = (const float*)d_in[0];
  const float* Wih = (const float*)d_in[1];
  const float* Whh = (const float*)d_in[2];
  const float* b   = (const float*)d_in[3];
  float* out = (float*)d_out;

  dim3 g1((SEQ * BATCH) / 128, HDIM / 64);
  rnn_proj_kernel<<<g1, 256, 0, stream>>>(x, Wih, b, out);
  rnn_scan_kernel<<<2, 1024, 0, stream>>>(Whh, out);
}
